// DIALModel_6786048328252
// MI455X (gfx1250) — matrix operand order repaired and checked by reading, not compiled, not run on hardware
//
#include <hip/hip_runtime.h>
#include <hip/hip_bf16.h>
#include <math.h>

typedef __bf16 bf16;
typedef __attribute__((ext_vector_type(16))) __bf16 v16bf;
typedef __attribute__((ext_vector_type(8)))  __bf16 v8bf;
typedef __attribute__((ext_vector_type(8)))  float  v8f;

#define NB   16
#define NN   512
#define DD   256
#define NHH  8
#define DHH  32
#define FFD  1024
#define EE   8192
#define TNE  544      /* padded ne-stage tokens (513 -> 17*32) */
#define TVNE 513
#define BCH  4        /* batch chunk for attention score buffers */

static __device__ __forceinline__ float blockReduceSum(float v, float* sh) {
  int tid = threadIdx.x;
  sh[tid] = v; __syncthreads();
  for (int s = blockDim.x >> 1; s > 0; s >>= 1) {
    if (tid < s) sh[tid] += sh[tid + s];
    __syncthreads();
  }
  float r = sh[0]; __syncthreads();
  return r;
}

static __device__ __forceinline__ float geluf(float x) {
  return 0.5f * x * (1.f + tanhf(0.7978845608028654f * (x + 0.044715f * x * x * x)));
}

// ---------------------------------------------------------------------------
// Generic batched bf16 WMMA GEMM: C[z] = A[z](MxK) * Bt[z](NxK)^T (+epilogue)
// One wave per (16*TM)x(16*TN) C tile; K step 32 (K must be multiple of 32).
// A row-major (lda), Bt stored N-major over K (ldb). f32 accumulate.
// Edge tiles: fragment-load rows/cols are clamped (EXEC stays all-ones for
// WMMA), stores are guarded. act: 0=none, 1=gelu, 2=silu.
// cMerge: scatter heads (z%8)*32 cols, (z/8)*mergeT rows.
// ---------------------------------------------------------------------------
template <int TM, int TN>
__global__ __launch_bounds__(32)
void k_gemm(const bf16* __restrict__ A, long sAz, int lda,
            const bf16* __restrict__ Bt, long sBz, int ldb,
            const float* __restrict__ biasN,
            const bf16* __restrict__ biasM, long sBiasZ, int ldBias, int biasDivNH,
            float* __restrict__ Cf, bf16* __restrict__ Cb, long sCz, int ldc,
            int cMerge, int mergeT, int M, int N, int K, int act)
{
  const int lane = threadIdx.x;
  const int z = blockIdx.z;
  const int M0 = blockIdx.x * (16 * TM), N0 = blockIdx.y * (16 * TN);
  const int mr = lane & 15, half = lane >> 4;
  const bf16* Az = A + (size_t)z * sAz;
  const bf16* Bz = Bt + (size_t)z * sBz;

  v8f zero8 = {0.f,0.f,0.f,0.f,0.f,0.f,0.f,0.f};
  v8f acc[TM][TN];
#pragma unroll
  for (int i = 0; i < TM; i++)
#pragma unroll
    for (int j = 0; j < TN; j++) acc[i][j] = zero8;

  for (int k0 = 0; k0 < K; k0 += 32) {
    v16bf af[TM];
#pragma unroll
    for (int i = 0; i < TM; i++) {
      // A fragment: row M0+i*16+mr ; lane-half selects K runs [h*8,+8) and [16+h*8,+8)
      int row = M0 + i * 16 + mr; if (row >= M) row = M - 1;
      const bf16* pa = Az + (size_t)row * lda + k0 + half * 8;
      v8bf lo = *(const v8bf*)pa;
      v8bf hi = *(const v8bf*)(pa + 16);
#pragma unroll
      for (int e = 0; e < 8; e++) { af[i][e] = lo[e]; af[i][e + 8] = hi[e]; }
      if (i == 0 && k0 + 32 < K) __builtin_prefetch(pa + 32, 0, 3);
    }
#pragma unroll
    for (int j = 0; j < TN; j++) {
      // B fragment: column N0+j*16+mr ; lane-half selects K runs [h*8,+8) and [16+h*8,+8)
      int col = N0 + j * 16 + mr; if (col >= N) col = N - 1;
      const bf16* pb = Bz + (size_t)col * ldb + k0 + half * 8;
      v8bf blo = *(const v8bf*)pb;
      v8bf bhi = *(const v8bf*)(pb + 16);
      v16bf bv;
#pragma unroll
      for (int e = 0; e < 8; e++) { bv[e] = blo[e]; bv[e + 8] = bhi[e]; }
      if (j == 0 && k0 + 32 < K) __builtin_prefetch(pb + 32, 0, 3);
#pragma unroll
      for (int i = 0; i < TM; i++)
        acc[i][j] = __builtin_amdgcn_wmma_f32_16x16x32_bf16(
            false, af[i], false, bv, (short)0, acc[i][j], false, false);
    }
  }

  size_t cbase = cMerge ? ((size_t)(z >> 3) * mergeT * ldc + (size_t)(z & 7) * DHH)
                        : (size_t)z * sCz;
#pragma unroll
  for (int j = 0; j < TN; j++) {
    int n = N0 + j * 16 + mr;
    if (n >= N) continue;
    float bn = biasN ? biasN[n] : 0.f;
    const bf16* bm = nullptr;
    if (biasM) {
      long bz = biasDivNH ? (z >> 3) : z;
      bm = biasM + (size_t)bz * sBiasZ + n;
    }
#pragma unroll
    for (int i = 0; i < TM; i++) {
#pragma unroll
      for (int r = 0; r < 8; r++) {
        int m = M0 + i * 16 + half * 8 + r;
        if (m >= M) continue;
        float v = acc[i][j][r] + bn;
        if (bm) v += (float)bm[(size_t)m * ldBias];
        if (act == 1) v = geluf(v);
        else if (act == 2) v = v * (1.f / (1.f + __expf(-v)));
        size_t off = cbase + (size_t)m * ldc + n;
        if (Cf) Cf[off] = v;
        if (Cb) Cb[off] = (bf16)v;
      }
    }
  }
}

// -------------------- small helper kernels --------------------
__global__ void k_a2bf(const float* __restrict__ s, bf16* __restrict__ d, long n) {
  long i = (long)blockIdx.x * 256 + threadIdx.x;
  if (i < n) d[i] = (bf16)s[i];
}

// W (KxN row-major f32) -> Wt (NxKpad bf16, K zero-padded)
__global__ void k_wconv(const float* __restrict__ W, bf16* __restrict__ Wt,
                        int K, int N, int Kpad) {
  long idx = (long)blockIdx.x * 256 + threadIdx.x;
  if (idx >= (long)N * Kpad) return;
  int k = (int)(idx % Kpad); long n = idx / Kpad;
  Wt[idx] = (k < K) ? (bf16)W[(size_t)k * N + n] : (bf16)0.f;
}

__global__ void k_ln_gelu(const float* __restrict__ X, const float* __restrict__ g,
                          const float* __restrict__ bb, bf16* __restrict__ out) {
  __shared__ float sh[256];
  int row = blockIdx.x, tid = threadIdx.x;
  const float* src = X + (size_t)row * 512;
  float x0 = src[tid], x1 = src[tid + 256];
  float mean = blockReduceSum(x0 + x1, sh) * (1.f / 512.f);
  float d0 = x0 - mean, d1 = x1 - mean;
  float var = blockReduceSum(d0 * d0 + d1 * d1, sh) * (1.f / 512.f);
  float inv = rsqrtf(var + 1e-5f);
  float y0 = d0 * inv * g[tid] + bb[tid];
  float y1 = d1 * inv * g[tid + 256] + bb[tid + 256];
  out[(size_t)row * 512 + tid]       = (bf16)geluf(y0);
  out[(size_t)row * 512 + tid + 256] = (bf16)geluf(y1);
}

// residual + layernorm over D=256 (row per block); optional bf16 mirror
__global__ void k_resln(const float* __restrict__ Hin, int inT, int inOff,
                        const float* __restrict__ X,
                        const float* __restrict__ g, const float* __restrict__ bb,
                        float* __restrict__ outF, bf16* __restrict__ outB, int T) {
  __shared__ float sh[256];
  int row = blockIdx.x, tid = threadIdx.x;
  int b = row / T, t = row - b * T;
  const float* src = Hin + ((size_t)b * inT + inOff + t) * DD;
  float x = src[tid];
  if (X) x += X[(size_t)row * DD + tid];
  float mean = blockReduceSum(x, sh) * (1.f / DD);
  float d = x - mean;
  float var = blockReduceSum(d * d, sh) * (1.f / DD);
  float y = d * rsqrtf(var + 1e-5f) * g[tid] + bb[tid];
  outF[(size_t)row * DD + tid] = y;
  if (outB) outB[(size_t)row * DD + tid] = (bf16)y;
}

__global__ void k_build_H(const float* __restrict__ G0, const int* __restrict__ indeg,
                          const float* __restrict__ deg_emb, const float* __restrict__ gtok,
                          float* __restrict__ H) {
  long idx = (long)blockIdx.x * 256 + threadIdx.x;
  if (idx >= (long)NB * TNE * DD) return;
  int d = (int)(idx & 255); long r = idx >> 8;
  int t = (int)(r % TNE); int b = (int)(r / TNE);
  float v;
  if (t == 0) v = gtok[d];
  else if (t <= NN) {
    int i = t - 1;
    int dg = indeg[b * NN + i]; dg = dg < 0 ? 0 : (dg > 511 ? 511 : dg);
    v = G0[((size_t)b * NN + i) * DD + d] + deg_emb[(size_t)dg * DD + d];
  } else v = 0.f;
  H[idx] = v;
}

// spatial + path bias -> (b,h,i,j) bf16, padded to TNE
__global__ void k_bias_ne(const int* __restrict__ dist, const float* __restrict__ pdat,
                          const float* __restrict__ pw, const float* __restrict__ semb,
                          const float* __restrict__ tvd, bf16* __restrict__ out) {
  long idx = (long)blockIdx.x * 256 + threadIdx.x;
  if (idx >= (long)NB * TNE * TNE) return;
  int j = (int)(idx % TNE); long r = idx / TNE;
  int i = (int)(r % TNE); int b = (int)(r / TNE);
  float val[NHH];
  if (i >= TVNE || j >= TVNE) {
    for (int h = 0; h < NHH; h++) val[h] = 0.f;
  } else if (i == 0 || j == 0) {
    for (int h = 0; h < NHH; h++) val[h] = tvd[h];
  } else {
    size_t e = ((size_t)b * NN + (i - 1)) * NN + (j - 1);
    int dv = dist[e];
    int dc = dv < 0 ? 0 : (dv > 510 ? 510 : dv);
    float plen = (float)(dv < 1 ? 1 : (dv > 5 ? 5 : dv));
    float pd[5];
    for (int l = 0; l < 5; l++) pd[l] = pdat[e * 5 + l];
    for (int h = 0; h < NHH; h++) {
      float pe = 0.f;
      for (int l = 0; l < 5; l++) pe += pd[l] * pw[l * NHH + h];
      val[h] = pe / plen + semb[(size_t)dc * NHH + h];
    }
  }
  for (int h = 0; h < NHH; h++)
    out[(((size_t)b * NHH + h) * TNE + i) * TNE + j] = (bf16)val[h];
}

// split QKV f32 planes -> head-major bf16 (Q scaled 1/sqrt(DH), V transposed)
__global__ void k_qkv_split(const float* __restrict__ QKV, long plane,
                            bf16* __restrict__ Qh, bf16* __restrict__ Kh,
                            bf16* __restrict__ Vt, int Tpad) {
  long idx = (long)blockIdx.x * 256 + threadIdx.x;
  if (idx >= (long)NB * NHH * Tpad * DHH) return;
  int dh = (int)(idx & 31); long r = idx >> 5;
  int t = (int)(r % Tpad); r /= Tpad;
  int h = (int)(r % NHH); int b = (int)(r / NHH);
  size_t src = ((size_t)b * Tpad + t) * DD + h * DHH + dh;
  Qh[idx] = (bf16)(QKV[src] * 0.17677669529663687f);
  Kh[idx] = (bf16)(QKV[plane + src]);
  Vt[((size_t)(b * NHH + h) * DHH + dh) * Tpad + t] = (bf16)(QKV[2 * plane + src]);
}

__global__ void k_softmax(const float* __restrict__ S, bf16* __restrict__ P,
                          int Tpad, int Tv) {
  __shared__ float sh[256];
  int row = blockIdx.x, tid = threadIdx.x;
  int i = row % Tpad;
  const float* sr = S + (size_t)row * Tpad;
  bf16* pr = P + (size_t)row * Tpad;
  if (i >= Tv) { for (int j = tid; j < Tpad; j += 256) pr[j] = (bf16)0.f; return; }
  float mx = -3.4e38f;
  for (int j = tid; j < Tv; j += 256) mx = fmaxf(mx, sr[j]);
  sh[tid] = mx; __syncthreads();
  for (int s = 128; s > 0; s >>= 1) { if (tid < s) sh[tid] = fmaxf(sh[tid], sh[tid + s]); __syncthreads(); }
  mx = sh[0]; __syncthreads();
  float sum = 0.f;
  for (int j = tid; j < Tv; j += 256) sum += __expf(sr[j] - mx);
  sum = blockReduceSum(sum, sh);
  float inv = 1.f / sum;
  for (int j = tid; j < Tpad; j += 256)
    pr[j] = (j < Tv) ? (bf16)(__expf(sr[j] - mx) * inv) : (bf16)0.f;
}

__global__ void k_init_gbias(bf16* __restrict__ gb) {
  long idx = (long)blockIdx.x * 256 + threadIdx.x;
  if (idx >= (long)NB * NN * NN) return;
  int j = (int)(idx & 511); int i = (int)((idx >> 9) & 511);
  gb[idx] = (bf16)((i == j) ? 0.f : -1e9f);
}

__global__ void k_edge_feat(int b, const int* __restrict__ ei, const float* __restrict__ H2,
                            const float* __restrict__ Sin, const float* __restrict__ Fin,
                            bf16* __restrict__ feat) {
  long idx = (long)blockIdx.x * 256 + threadIdx.x;
  if (idx >= (long)EE * 544) return;
  int k = (int)(idx % 544); long e = idx / 544;
  int u = ei[(size_t)b * 2 * EE + e];
  int v = ei[(size_t)b * 2 * EE + EE + e];
  float val;
  if (k < 256) {
    float a = H2[((size_t)b * NN + u) * DD + k], c = H2[((size_t)b * NN + v) * DD + k];
    val = fabsf(a - c);
  } else if (k < 512) {
    int kk = k - 256;
    float a = H2[((size_t)b * NN + u) * DD + kk], c = H2[((size_t)b * NN + v) * DD + kk];
    val = a * c;
  } else if (k == 512) val = Sin[((size_t)b * NN + u) * NN + v];
  else if (k == 513) val = Fin[((size_t)b * NN + u) * NN + v];
  else val = 0.f;
  feat[idx] = (bf16)val;
}

__global__ void k_edge_energy(int b, const int* __restrict__ ei, const bf16* __restrict__ hid,
                              const float* __restrict__ W2, const float* __restrict__ b2,
                              const float* __restrict__ thr, const float* __restrict__ Sin,
                              bf16* __restrict__ gb) {
  __shared__ float sh[256];
  int e = blockIdx.x, tid = threadIdx.x;
  float a = 0.f;
  for (int j = tid; j < 512; j += 256) a += (float)hid[(size_t)e * 512 + j] * W2[j];
  float en = blockReduceSum(a, sh);
  if (tid == 0) {
    en += b2[0];
    int u = ei[(size_t)b * 2 * EE + e];
    int v = ei[(size_t)b * 2 * EE + EE + e];
    float Se = Sin[((size_t)b * NN + u) * NN + v];
    float m = 1.f / (1.f + __expf(-((en - thr[0] - Se) / 8.f)));
    float lm = __logf(m + 1e-9f);
    gb[(size_t)b * NN * NN + (size_t)u * NN + v] = (bf16)lm;
    gb[(size_t)b * NN * NN + (size_t)v * NN + u] = (bf16)lm;
  }
}

__global__ void k_mean(const float* __restrict__ Z, float* __restrict__ g) {
  int b = blockIdx.x, d = threadIdx.x;
  float s = 0.f;
  for (int t = 0; t < NN; t++) s += Z[((size_t)b * NN + t) * DD + d];
  g[b * DD + d] = s * (1.f / NN);
}

__global__ void k_head(const float* __restrict__ g, const float* __restrict__ W1,
                       const float* __restrict__ b1, const float* __restrict__ W2,
                       const float* __restrict__ b2, float* __restrict__ out) {
  __shared__ float gs[256];
  __shared__ float sh[512];
  int b = blockIdx.x, j = threadIdx.x;
  if (j < 256) gs[j] = g[b * DD + j];
  __syncthreads();
  float acc = b1[j];
  for (int k = 0; k < 256; k++) acc += gs[k] * W1[(size_t)k * 512 + j];
  float h = fmaxf(acc, 0.f);
  for (int o = 0; o < 2; o++) {
    sh[j] = h * W2[(size_t)j * 2 + o]; __syncthreads();
    for (int s = 256; s > 0; s >>= 1) { if (j < s) sh[j] += sh[j + s]; __syncthreads(); }
    if (j == 0) out[b * 2 + o] = sh[0] + b2[o];
    __syncthreads();
  }
}

// ---------------------------------------------------------------------------
extern "C" void kernel_launch(void* const* d_in, const int* in_sizes, int n_in,
                              void* d_out, int out_size, void* d_ws, size_t ws_size,
                              hipStream_t stream)
{
  (void)in_sizes; (void)n_in; (void)out_size; (void)ws_size;
  auto F = [&](int i) { return (const float*)d_in[i]; };
  auto I = [&](int i) { return (const int*)d_in[i]; };
  char* ws = (char*)d_ws;
  size_t off = 0;
  auto alloc = [&](size_t bytes) -> char* {
    char* p = ws + off;
    off = (off + bytes + 255) & ~(size_t)255;
    return p;
  };
  auto cdiv = [](long a, long b) { return (unsigned)((a + b - 1) / b); };

  // ---- bf16 weight staging (N-major over padded K) ----
  bf16* WprojW1 = (bf16*)alloc((size_t)512 * 512 * 2);
  bf16* WprojW2 = (bf16*)alloc((size_t)256 * 512 * 2);
  bf16* Wqkvo[2][2][4]; bf16* Wff1[2][2]; bf16* Wff2[2][2];
  for (int s = 0; s < 2; s++)
    for (int l = 0; l < 2; l++) {
      for (int m = 0; m < 4; m++) Wqkvo[s][l][m] = (bf16*)alloc((size_t)256 * 256 * 2);
      Wff1[s][l] = (bf16*)alloc((size_t)1024 * 256 * 2);
      Wff2[s][l] = (bf16*)alloc((size_t)256 * 1024 * 2);
    }
  bf16* Weg1 = (bf16*)alloc((size_t)512 * 544 * 2);

  // ---- activation / attention workspace ----
  bf16*  XA   = (bf16*)alloc((size_t)8192 * 512 * 2);          // bf16 A staging
  bf16*  XB   = (bf16*)alloc((size_t)(NB * TNE) * 1024 * 2);   // FF mid / edge feat
  float* X1f  = (float*)alloc((size_t)8192 * 512 * 4);
  float* QKVf = (float*)alloc((size_t)3 * NB * TNE * 256 * 4);
  float* Hf   = (float*)alloc((size_t)NB * TNE * 256 * 4);
  float* Htmp = (float*)alloc((size_t)NB * TNE * 256 * 4);
  float* AOut = (float*)alloc((size_t)NB * TNE * 256 * 4);
  float* H2f  = (float*)alloc((size_t)NB * NN * 256 * 4);
  bf16*  Qh   = (bf16*)alloc((size_t)NB * NHH * TNE * DHH * 2);
  bf16*  Kh   = (bf16*)alloc((size_t)NB * NHH * TNE * DHH * 2);
  bf16*  Vt   = (bf16*)alloc((size_t)NB * NHH * TNE * DHH * 2);
  bf16*  BiasNE = (bf16*)alloc((size_t)NB * NHH * TNE * TNE * 2);
  float* Sbuf = (float*)alloc((size_t)BCH * NHH * TNE * TNE * 4);
  bf16*  Pbuf = (bf16*)alloc((size_t)BCH * NHH * TNE * TNE * 2);
  bf16*  Gbias = (bf16*)alloc((size_t)NB * NN * NN * 2);
  float* Gvec = (float*)alloc((size_t)NB * 256 * 4);

  auto wconv = [&](const float* W, bf16* Wt, int K, int N, int Kpad) {
    long n = (long)N * Kpad;
    k_wconv<<<cdiv(n, 256), 256, 0, stream>>>(W, Wt, K, N, Kpad);
  };
  auto gemm = [&](const bf16* A, long sAz, int lda, const bf16* Bt, long sBz, int ldb,
                  const float* biasN, const bf16* biasM, long sBiasZ, int ldBias, int divNH,
                  float* Cf, bf16* Cb, long sCz, int ldc, int cMerge, int mergeT,
                  int M, int N, int K, int Z, int act) {
    if (N <= 32) {
      dim3 g(cdiv(M, 64), cdiv(N, 16), (unsigned)Z);
      k_gemm<4,1><<<g, 32, 0, stream>>>(A, sAz, lda, Bt, sBz, ldb, biasN, biasM, sBiasZ,
                                        ldBias, divNH, Cf, Cb, sCz, ldc, cMerge, mergeT,
                                        M, N, K, act);
    } else {
      dim3 g(cdiv(M, 64), cdiv(N, 64), (unsigned)Z);
      k_gemm<4,4><<<g, 32, 0, stream>>>(A, sAz, lda, Bt, sBz, ldb, biasN, biasM, sBiasZ,
                                        ldBias, divNH, Cf, Cb, sCz, ldc, cMerge, mergeT,
                                        M, N, K, act);
    }
  };

  // ---- weight prep ----
  const int wIdx[2] = {19, 37};  // ne_Wq, gt_Wq input slots
  wconv(F(8), WprojW1, 512, 512, 512);
  wconv(F(12), WprojW2, 512, 256, 512);
  for (int s = 0; s < 2; s++)
    for (int l = 0; l < 2; l++) {
      int base = wIdx[s];
      for (int m = 0; m < 4; m++)
        wconv(F(base + 2 * m) + (size_t)l * 256 * 256, Wqkvo[s][l][m], 256, 256, 256);
      wconv(F(base + 10) + (size_t)l * 256 * 1024, Wff1[s][l], 256, 1024, 256);
      wconv(F(base + 12) + (size_t)l * 1024 * 256, Wff2[s][l], 1024, 256, 1024);
    }
  wconv(F(55), Weg1, 514, 512, 544);

  // ---- input projection ----
  { long n = (long)8192 * 512; k_a2bf<<<cdiv(n,256),256,0,stream>>>(F(0), XA, n); }
  gemm(XA,0,512, WprojW1,0,512, F(9), nullptr,0,0,0, X1f,nullptr,0,512, 0,0, 8192,512,512, 1, 0);
  k_ln_gelu<<<8192, 256, 0, stream>>>(X1f, F(10), F(11), XA);
  gemm(XA,0,512, WprojW2,0,512, F(13), nullptr,0,0,0, QKVf,nullptr,0,256, 0,0, 8192,256,512, 1, 0);
  { long n = (long)NB * TNE * 256; k_build_H<<<cdiv(n,256),256,0,stream>>>(QKVf, I(1), F(14), F(17), Hf); }
  { long n = (long)NB * TNE * TNE; k_bias_ne<<<cdiv(n,256),256,0,stream>>>(I(4), F(3), F(16), F(15), F(18), BiasNE); }

  // ---- transformer stage (shared by ne and gt) ----
  auto xformer = [&](int s, float* Hbase, int Tpad, int Tv,
                     const bf16* biasMat, long biasZstride, int biasDiv, long biasBstride) {
    int M = NB * Tpad;
    long plane = (long)M * 256;
    int base = wIdx[s];
    for (int l = 0; l < 2; l++) {
      { long n = (long)M * 256; k_a2bf<<<cdiv(n,256),256,0,stream>>>(Hbase, XA, n); }
      gemm(XA,0,256, Wqkvo[s][l][0],0,256, F(base+1)+(size_t)l*256, nullptr,0,0,0,
           QKVf,            nullptr,0,256, 0,0, M,256,256, 1, 0);
      gemm(XA,0,256, Wqkvo[s][l][1],0,256, F(base+3)+(size_t)l*256, nullptr,0,0,0,
           QKVf+plane,      nullptr,0,256, 0,0, M,256,256, 1, 0);
      gemm(XA,0,256, Wqkvo[s][l][2],0,256, F(base+5)+(size_t)l*256, nullptr,0,0,0,
           QKVf+2*plane,    nullptr,0,256, 0,0, M,256,256, 1, 0);
      { long n = (long)NB * NHH * Tpad * DHH;
        k_qkv_split<<<cdiv(n,256),256,0,stream>>>(QKVf, plane, Qh, Kh, Vt, Tpad); }
      for (int c = 0; c < NB / BCH; c++) {
        long qoff = (long)c * BCH * NHH * Tpad * DHH;
        // S = Q K^T (+bias)
        gemm(Qh+qoff,(long)Tpad*DHH,DHH, Kh+qoff,(long)Tpad*DHH,DHH, nullptr,
             biasMat + (size_t)c * BCH * biasBstride, biasZstride, Tpad, biasDiv,
             Sbuf,nullptr,(long)Tpad*Tpad,Tpad, 0,0, Tpad,Tpad,DHH, BCH*NHH, 0);
        k_softmax<<<(unsigned)(BCH*NHH*Tpad), 256, 0, stream>>>(Sbuf, Pbuf, Tpad, Tv);
        // O = P V (heads merged into (b,t,d))
        gemm(Pbuf,(long)Tpad*Tpad,Tpad, Vt+qoff,(long)DHH*Tpad,Tpad, nullptr,
             nullptr,0,0,0,
             Htmp + (size_t)c*BCH*Tpad*256, nullptr, 0, 256, 1, Tpad,
             Tpad,DHH,Tpad, BCH*NHH, 0);
      }
      { long n = (long)M * 256; k_a2bf<<<cdiv(n,256),256,0,stream>>>(Htmp, XA, n); }
      gemm(XA,0,256, Wqkvo[s][l][3],0,256, F(base+7)+(size_t)l*256, nullptr,0,0,0,
           AOut,nullptr,0,256, 0,0, M,256,256, 1, 0);
      k_resln<<<(unsigned)M,256,0,stream>>>(Hbase, Tpad, 0, AOut,
          F(base+8)+(size_t)l*256, F(base+9)+(size_t)l*256, Hbase, XA, Tpad);
      // FFN
      gemm(XA,0,256, Wff1[s][l],0,256, F(base+11)+(size_t)l*1024, nullptr,0,0,0,
           nullptr, XB, 0,1024, 0,0, M,1024,256, 1, 1 /*gelu*/);
      gemm(XB,0,1024, Wff2[s][l],0,1024, F(base+13)+(size_t)l*256, nullptr,0,0,0,
           AOut,nullptr,0,256, 0,0, M,256,1024, 1, 0);
      k_resln<<<(unsigned)M,256,0,stream>>>(Hbase, Tpad, 0, AOut,
          F(base+14)+(size_t)l*256, F(base+15)+(size_t)l*256, Hbase, XA, Tpad);
    }
  };

  // ---- node-encoder layers (T=513 padded to 544) ----
  xformer(0, Hf, TNE, TVNE, BiasNE, (long)TNE*TNE, 0, (long)NHH*TNE*TNE);
  // ne_norm: drop graph token, LN -> H2
  k_resln<<<(unsigned)(NB*NN),256,0,stream>>>(Hf, TNE, 1, nullptr, F(35), F(36), H2f, XA, NN);

  // ---- edge gating -> gbias ----
  { long n = (long)NB * NN * NN; k_init_gbias<<<cdiv(n,256),256,0,stream>>>(Gbias); }
  for (int b = 0; b < NB; b++) {
    { long n = (long)EE * 544; k_edge_feat<<<cdiv(n,256),256,0,stream>>>(b, I(7), H2f, F(5), F(6), XB); }
    gemm(XB,0,544, Weg1,0,544, F(56), nullptr,0,0,0,
         nullptr, XA, 0,512, 0,0, EE,512,544, 1, 2 /*silu*/);
    k_edge_energy<<<EE,256,0,stream>>>(b, I(7), XA, F(57), F(58), F(59), F(5), Gbias);
  }

  // ---- graph-transformer layers (T=512) ----
  xformer(1, H2f, NN, NN, Gbias, (long)NN*NN, 1, (long)NN*NN);

  // ---- final norm, pool, head ----
  k_resln<<<(unsigned)(NB*NN),256,0,stream>>>(H2f, NN, 0, nullptr, F(53), F(54), AOut, nullptr, NN);
  k_mean<<<NB,256,0,stream>>>(AOut, Gvec);
  k_head<<<NB,512,0,stream>>>(Gvec, F(60), F(61), F(62), F(63), (float*)d_out);
}
